// LSTMEnvGoal_90108413870860
// MI455X (gfx1250) — matrix operand order repaired and checked by reading, not compiled, not run on hardware
//
#include <hip/hip_runtime.h>

#define HDs   256
#define TENC  120
#define TDEC  360
#define NB    512

typedef __attribute__((ext_vector_type(16))) _Float16 v16h;
typedef __attribute__((ext_vector_type(8)))  _Float16 v8h;
typedef __attribute__((ext_vector_type(8)))  float    v8f;

__device__ __forceinline__ float sigf(float x) { return 1.0f / (1.0f + __expf(-x)); }

// A-fragment (16x32 f16, MxK). ISA 7.12.2: lanes 0-15 row=lane K=0..7 / 16..23;
// lanes 16-31 row=lane-16 K=8..15 / 24..31. base points at [row0, k0].
__device__ __forceinline__ v16h load_a_frag(const _Float16* base, int lane, int stride) {
    const int r = lane & 15;
    const _Float16* q = base + r * stride + ((lane & 16) ? 8 : 0);
    v8h lo = *(const v8h*)(q);
    v8h hi = *(const v8h*)(q + 16);
    v16h a;
#pragma unroll
    for (int i = 0; i < 8; ++i) { a[i] = lo[i]; a[i + 8] = hi[i]; }
    return a;
}

// B-fragment (32x16 f16, KxN), weight stored row-major [N(out) x K(in)], stride=K.
// lanes 0-15: col=lane, K=0..7 / 16..23; lanes 16-31: col=lane-16, K=8..15 / 24..31.
__device__ __forceinline__ v16h load_b_frag(const _Float16* base, int lane, int stride) {
    const int n  = lane & 15;
    const int kb = (lane & 16) ? 8 : 0;
    const _Float16* q = base + n * stride + kb;
    v8h lo = *(const v8h*)(q);
    v8h hi = *(const v8h*)(q + 16);
    v16h b;
#pragma unroll
    for (int i = 0; i < 8; ++i) { b[i] = lo[i]; b[i + 8] = hi[i]; }
    return b;
}

__device__ __forceinline__ v8f wmma_acc(v16h a, v16h b, v8f c) {
    return __builtin_amdgcn_wmma_f32_16x16x32_f16(false, a, false, b, (short)0, c, false, false);
}

// K=256 GEMM-accumulate. unroll-2 body: the compiler clauses the 2x(2 global +
// 2 LDS) loads ahead of both WMMAs and uses split waits, overlapping the second
// WMMA's loads with the first WMMA. Deeper unrolling / manual pipelining causes
// B-fragment spills to scratch (verified in earlier rounds) - keep this shape.
__device__ __forceinline__ v8f gemm256(const _Float16* abase, const _Float16* wbase,
                                       int lane, v8f acc) {
#pragma unroll 2
    for (int k = 0; k < HDs; k += 32) {
        v16h af = load_a_frag(abase + k, lane, HDs);
        v16h bf = load_b_frag(wbase + k, lane, HDs);
        acc = wmma_acc(af, bf, acc);
    }
    return acc;
}

// ---------------- weight conversion ----------------
__global__ void f32tof16_kernel(const float* __restrict__ src, _Float16* __restrict__ dst, int n) {
    int i = blockIdx.x * blockDim.x + threadIdx.x;
    if (i < n) dst[i] = (_Float16)src[i];
}

// dec_Wih0 [1024,258] -> ctx part [1024,256] (cols 2..257) + cur part [1024,2]
__global__ void split_wih0_kernel(const float* __restrict__ src, _Float16* __restrict__ wctx,
                                  _Float16* __restrict__ wcur) {
    int i = blockIdx.x * blockDim.x + threadIdx.x;
    if (i >= 1024 * 258) return;
    int r = i / 258, c = i % 258;
    if (c < 2) wcur[r * 2 + c] = (_Float16)src[i];
    else       wctx[r * 256 + (c - 2)] = (_Float16)src[i];
}

// ---------------- encoder LSTM (persistent per 16-row batch tile) ----------------
__global__ __launch_bounds__(256) void encoder_kernel(
    const float* __restrict__ hist, const float* __restrict__ Wih0,
    const float* __restrict__ bih0, const float* __restrict__ bhh0,
    const float* __restrict__ bih1, const float* __restrict__ bhh1,
    const _Float16* __restrict__ Whh0, const _Float16* __restrict__ Wih1,
    const _Float16* __restrict__ Whh1,
    float* __restrict__ h0o, float* __restrict__ c0o,
    float* __restrict__ h1o, float* __restrict__ c1o) {
    __shared__ _Float16 h0buf[2][16 * HDs];
    __shared__ _Float16 h1buf[2][16 * HDs];
    __shared__ float xs[32];
    const int tid = threadIdx.x, wv = tid >> 5, lane = tid & 31;
    const int n = lane & 15, mb = (lane & 16) ? 8 : 0;
    const int row0 = blockIdx.x * 16;

    for (int i = tid; i < 16 * HDs; i += 256) {
        h0buf[1][i] = (_Float16)0.f;
        h1buf[1][i] = (_Float16)0.f;
    }
    float b0c[2][4], b1c[2][4], wx0c[2][4], wx1c[2][4];
#pragma unroll
    for (int s = 0; s < 2; ++s)
#pragma unroll
        for (int g = 0; g < 4; ++g) {
            int col = g * 256 + wv * 32 + s * 16 + n;
            b0c[s][g] = bih0[col] + bhh0[col];
            b1c[s][g] = bih1[col] + bhh1[col];
            wx0c[s][g] = Wih0[col * 2 + 0];
            wx1c[s][g] = Wih0[col * 2 + 1];
        }
    v8f c0r[2], c1r[2];
#pragma unroll
    for (int s = 0; s < 2; ++s)
#pragma unroll
        for (int r = 0; r < 8; ++r) { c0r[s][r] = 0.f; c1r[s][r] = 0.f; }
    __syncthreads();

    for (int t = 0; t < TENC; ++t) {
        const int p = t & 1, q = p ^ 1;
        if (tid < 32) xs[tid] = hist[((size_t)(row0 + (tid >> 1)) * TENC + t) * 2 + (tid & 1)];
        __syncthreads();
        float xr0[8], xr1[8];
#pragma unroll
        for (int r = 0; r < 8; ++r) { xr0[r] = xs[(mb + r) * 2]; xr1[r] = xs[(mb + r) * 2 + 1]; }

        // layer 0: g = x@Wih0^T + h0@Whh0^T + b
#pragma unroll
        for (int s = 0; s < 2; ++s) {
            v8f acc[4];
#pragma unroll
            for (int g = 0; g < 4; ++g) {
                v8f a;
#pragma unroll
                for (int r = 0; r < 8; ++r)
                    a[r] = b0c[s][g] + xr0[r] * wx0c[s][g] + xr1[r] * wx1c[s][g];
                const int col0 = g * 256 + wv * 32 + s * 16;
                acc[g] = gemm256(&h0buf[q][0], Whh0 + (size_t)col0 * HDs, lane, a);
            }
            const int u0 = wv * 32 + s * 16 + n;
#pragma unroll
            for (int r = 0; r < 8; ++r) {
                float iv = sigf(acc[0][r]), fv = sigf(acc[1][r]);
                float gv = tanhf(acc[2][r]), ov = sigf(acc[3][r]);
                float cv = fv * c0r[s][r] + iv * gv; c0r[s][r] = cv;
                float hv = ov * tanhf(cv);
                h0buf[p][(mb + r) * HDs + u0] = (_Float16)hv;
                if (t == TENC - 1) {
                    h0o[(size_t)(row0 + mb + r) * HDs + u0] = hv;
                    c0o[(size_t)(row0 + mb + r) * HDs + u0] = cv;
                }
            }
        }
        __syncthreads();
        // layer 1: g = h0@Wih1^T + h1@Whh1^T + b
#pragma unroll
        for (int s = 0; s < 2; ++s) {
            v8f acc[4];
#pragma unroll
            for (int g = 0; g < 4; ++g) {
                v8f a;
#pragma unroll
                for (int r = 0; r < 8; ++r) a[r] = b1c[s][g];
                const int col0 = g * 256 + wv * 32 + s * 16;
                a = gemm256(&h0buf[p][0], Wih1 + (size_t)col0 * HDs, lane, a);
                acc[g] = gemm256(&h1buf[q][0], Whh1 + (size_t)col0 * HDs, lane, a);
            }
            const int u0 = wv * 32 + s * 16 + n;
#pragma unroll
            for (int r = 0; r < 8; ++r) {
                float iv = sigf(acc[0][r]), fv = sigf(acc[1][r]);
                float gv = tanhf(acc[2][r]), ov = sigf(acc[3][r]);
                float cv = fv * c1r[s][r] + iv * gv; c1r[s][r] = cv;
                float hv = ov * tanhf(cv);
                h1buf[p][(mb + r) * HDs + u0] = (_Float16)hv;
                if (t == TENC - 1) {
                    h1o[(size_t)(row0 + mb + r) * HDs + u0] = hv;
                    c1o[(size_t)(row0 + mb + r) * HDs + u0] = cv;
                }
            }
        }
        __syncthreads();
    }
}

// ---------------- decoder LSTM ----------------
__global__ __launch_bounds__(256) void decoder_kernel(
    const float* __restrict__ hist,
    const float* __restrict__ h0i, const float* __restrict__ c0i,
    const float* __restrict__ c1i, const float* __restrict__ ctx,
    const float* __restrict__ bih0, const float* __restrict__ bhh0,
    const float* __restrict__ bih1, const float* __restrict__ bhh1,
    const _Float16* __restrict__ Wctx, const _Float16* __restrict__ Wcur,
    const _Float16* __restrict__ Whh0, const _Float16* __restrict__ Wih1,
    const _Float16* __restrict__ Whh1,
    const float* __restrict__ outw, const float* __restrict__ outb,
    float* __restrict__ out) {
    __shared__ _Float16 h0buf[2][16 * HDs];
    __shared__ _Float16 h1buf[2][16 * HDs];
    __shared__ float currs[32];
    const int tid = threadIdx.x, wv = tid >> 5, lane = tid & 31;
    const int n = lane & 15, mb = (lane & 16) ? 8 : 0;
    const int row0 = blockIdx.x * 16;

    for (int i = tid; i < 16 * HDs; i += 256) {
        h0buf[1][i] = (_Float16)h0i[(size_t)row0 * HDs + i];
        h1buf[1][i] = (_Float16)ctx[(size_t)row0 * HDs + i];  // h1 init = context
    }
    if (tid < 32)
        currs[tid] = hist[((size_t)(row0 + (tid >> 1)) * TENC + (TENC - 1)) * 2 + (tid & 1)];

    float b1c[2][4], wc0[2][4], wc1[2][4];
#pragma unroll
    for (int s = 0; s < 2; ++s)
#pragma unroll
        for (int g = 0; g < 4; ++g) {
            int col = g * 256 + wv * 32 + s * 16 + n;
            b1c[s][g] = bih1[col] + bhh1[col];
            wc0[s][g] = (float)Wcur[col * 2 + 0];
            wc1[s][g] = (float)Wcur[col * 2 + 1];
        }
    v8f c0r[2], c1r[2];
#pragma unroll
    for (int s = 0; s < 2; ++s)
#pragma unroll
        for (int r = 0; r < 8; ++r) {
            int u = wv * 32 + s * 16 + n;
            c0r[s][r] = c0i[(size_t)(row0 + mb + r) * HDs + u];
            c1r[s][r] = c1i[(size_t)(row0 + mb + r) * HDs + u];
        }
    __syncthreads();

    // Constant across all 360 steps: context @ Wih0[:,2:258]^T + bih0 + bhh0.
    // Held in f16 registers (v8h) to keep total VGPR usage under control.
    v8h ctxch[2][4];
#pragma unroll
    for (int s = 0; s < 2; ++s)
#pragma unroll
        for (int g = 0; g < 4; ++g) {
            int col = g * 256 + wv * 32 + s * 16 + n;
            v8f a;
            float bc = bih0[col] + bhh0[col];
#pragma unroll
            for (int r = 0; r < 8; ++r) a[r] = bc;
            const int col0 = g * 256 + wv * 32 + s * 16;
            a = gemm256(&h1buf[1][0], Wctx + (size_t)col0 * HDs, lane, a);  // context (f16)
#pragma unroll
            for (int r = 0; r < 8; ++r) ctxch[s][g][r] = (_Float16)a[r];
        }

    for (int t = 0; t < TDEC; ++t) {
        const int p = t & 1, q = p ^ 1;
        float cr0[8], cr1[8];
#pragma unroll
        for (int r = 0; r < 8; ++r) { cr0[r] = currs[(mb + r) * 2]; cr1[r] = currs[(mb + r) * 2 + 1]; }

        // layer 0
#pragma unroll
        for (int s = 0; s < 2; ++s) {
            v8f acc[4];
#pragma unroll
            for (int g = 0; g < 4; ++g) {
                v8f a;
#pragma unroll
                for (int r = 0; r < 8; ++r)
                    a[r] = (float)ctxch[s][g][r] + cr0[r] * wc0[s][g] + cr1[r] * wc1[s][g];
                const int col0 = g * 256 + wv * 32 + s * 16;
                acc[g] = gemm256(&h0buf[q][0], Whh0 + (size_t)col0 * HDs, lane, a);
            }
            const int u0 = wv * 32 + s * 16 + n;
#pragma unroll
            for (int r = 0; r < 8; ++r) {
                float iv = sigf(acc[0][r]), fv = sigf(acc[1][r]);
                float gv = tanhf(acc[2][r]), ov = sigf(acc[3][r]);
                float cv = fv * c0r[s][r] + iv * gv; c0r[s][r] = cv;
                h0buf[p][(mb + r) * HDs + u0] = (_Float16)(ov * tanhf(cv));
            }
        }
        __syncthreads();
        // layer 1
#pragma unroll
        for (int s = 0; s < 2; ++s) {
            v8f acc[4];
#pragma unroll
            for (int g = 0; g < 4; ++g) {
                v8f a;
#pragma unroll
                for (int r = 0; r < 8; ++r) a[r] = b1c[s][g];
                const int col0 = g * 256 + wv * 32 + s * 16;
                a = gemm256(&h0buf[p][0], Wih1 + (size_t)col0 * HDs, lane, a);
                acc[g] = gemm256(&h1buf[q][0], Whh1 + (size_t)col0 * HDs, lane, a);
            }
            const int u0 = wv * 32 + s * 16 + n;
#pragma unroll
            for (int r = 0; r < 8; ++r) {
                float iv = sigf(acc[0][r]), fv = sigf(acc[1][r]);
                float gv = tanhf(acc[2][r]), ov = sigf(acc[3][r]);
                float cv = fv * c1r[s][r] + iv * gv; c1r[s][r] = cv;
                h1buf[p][(mb + r) * HDs + u0] = (_Float16)(ov * tanhf(cv));
            }
        }
        __syncthreads();
        // output head: curr += h1 @ out_w^T + out_b  (all 8 waves cooperate)
        {
            const int pair = wv * 4 + (lane >> 3);  // 0..31 : (row 0..15, od 0..1)
            const int seg  = lane & 7;
            const int prow = pair >> 1, od = pair & 1;
            const _Float16* hp = &h1buf[p][prow * HDs + seg * 32];
            const float* wp = outw + od * HDs + seg * 32;
            float psum = 0.f;
#pragma unroll
            for (int k2 = 0; k2 < 32; ++k2) psum += (float)hp[k2] * wp[k2];
            psum += __shfl_xor(psum, 4, 32);
            psum += __shfl_xor(psum, 2, 32);
            psum += __shfl_xor(psum, 1, 32);
            if (seg == 0) {
                float nc = currs[pair] + psum + outb[od];
                currs[pair] = nc;
                out[((size_t)(row0 + prow) * TDEC + t) * 2 + od] = nc;
            }
        }
        __syncthreads();
    }
}

// ---------------- CNN / FC path (small fraction of FLOPs) ----------------
__global__ void conv_s2_kernel(const float* __restrict__ in, const float* __restrict__ w,
                               const float* __restrict__ b, float* __restrict__ out,
                               int N, int Cin, int Hin, int Cout, int Hout) {
    int idx = blockIdx.x * blockDim.x + threadIdx.x;
    int total = N * Cout * Hout * Hout;
    if (idx >= total) return;
    int ox = idx % Hout; int tmp = idx / Hout;
    int oy = tmp % Hout; tmp /= Hout;
    int co = tmp % Cout; int img = tmp / Cout;
    float acc = b[co];
    const float* wp = w + (size_t)co * Cin * 9;
    const float* ip = in + (size_t)img * Cin * Hin * Hin;
    int iy0 = oy * 2 - 1, ix0 = ox * 2 - 1;
    for (int ci = 0; ci < Cin; ++ci) {
        for (int ky = 0; ky < 3; ++ky) {
            int iy = iy0 + ky;
            if (iy < 0 || iy >= Hin) continue;
            for (int kx = 0; kx < 3; ++kx) {
                int ix = ix0 + kx;
                if (ix < 0 || ix >= Hin) continue;
                acc += ip[(size_t)ci * Hin * Hin + iy * Hin + ix] * wp[ci * 9 + ky * 3 + kx];
            }
        }
    }
    out[idx] = fmaxf(acc, 0.f);
}

__global__ void pool_kernel(const float* __restrict__ e3, float* __restrict__ pooled) {
    int idx = blockIdx.x * blockDim.x + threadIdx.x;  // [512*128]
    if (idx >= NB * 128) return;
    const float* p = e3 + (size_t)idx * 64;
    float s = 0.f;
    for (int i = 0; i < 64; ++i) s += p[i];
    pooled[idx] = s * (1.0f / 64.0f);
}

__global__ void fc_kernel(const float* __restrict__ x, const float* __restrict__ w,
                          const float* __restrict__ b, float* __restrict__ y,
                          int B, int IN, int OUT, int relu) {
    int idx = blockIdx.x * blockDim.x + threadIdx.x;
    if (idx >= B * OUT) return;
    int o = idx % OUT, bi = idx / OUT;
    float acc = b[o];
    const float* xp = x + (size_t)bi * IN;
    const float* wp = w + (size_t)o * IN;
    for (int k = 0; k < IN; ++k) acc += xp[k] * wp[k];
    y[idx] = relu ? fmaxf(acc, 0.f) : acc;
}

__global__ void fusion_kernel(const float* __restrict__ h1, const float* __restrict__ env,
                              const float* __restrict__ goal, const float* __restrict__ w,
                              const float* __restrict__ b, float* __restrict__ ctx) {
    int idx = blockIdx.x * blockDim.x + threadIdx.x;  // [512*256]
    if (idx >= NB * HDs) return;
    int o = idx % HDs, bi = idx / HDs;
    const float* wp = w + (size_t)o * 448;
    float acc = b[o];
    const float* hp = h1 + (size_t)bi * HDs;
    for (int k = 0; k < HDs; ++k) acc += hp[k] * wp[k];
    const float* ep = env + (size_t)bi * 128;
    for (int k = 0; k < 128; ++k) acc += ep[k] * wp[256 + k];
    const float* gp = goal + (size_t)bi * 64;
    for (int k = 0; k < 64; ++k) acc += gp[k] * wp[384 + k];
    ctx[idx] = fmaxf(acc, 0.f);
}

extern "C" void kernel_launch(void* const* d_in, const int* in_sizes, int n_in,
                              void* d_out, int out_size, void* d_ws, size_t ws_size,
                              hipStream_t stream) {
    const float* hist     = (const float*)d_in[0];
    const float* env_map  = (const float*)d_in[1];
    const float* goal     = (const float*)d_in[2];
    const float* eWih0    = (const float*)d_in[3];
    const float* eWhh0    = (const float*)d_in[4];
    const float* ebih0    = (const float*)d_in[5];
    const float* ebhh0    = (const float*)d_in[6];
    const float* eWih1    = (const float*)d_in[7];
    const float* eWhh1    = (const float*)d_in[8];
    const float* ebih1    = (const float*)d_in[9];
    const float* ebhh1    = (const float*)d_in[10];
    const float* c1w = (const float*)d_in[11]; const float* c1b = (const float*)d_in[12];
    const float* c2w = (const float*)d_in[13]; const float* c2b = (const float*)d_in[14];
    const float* c3w = (const float*)d_in[15]; const float* c3b = (const float*)d_in[16];
    const float* efcw = (const float*)d_in[17]; const float* efcb = (const float*)d_in[18];
    const float* gfcw = (const float*)d_in[19]; const float* gfcb = (const float*)d_in[20];
    const float* fw   = (const float*)d_in[21]; const float* fb   = (const float*)d_in[22];
    const float* dWih0 = (const float*)d_in[23];
    const float* dWhh0 = (const float*)d_in[24];
    const float* dbih0 = (const float*)d_in[25];
    const float* dbhh0 = (const float*)d_in[26];
    const float* dWih1 = (const float*)d_in[27];
    const float* dWhh1 = (const float*)d_in[28];
    const float* dbih1 = (const float*)d_in[29];
    const float* dbhh1 = (const float*)d_in[30];
    const float* outw = (const float*)d_in[31];
    const float* outb = (const float*)d_in[32];
    float* out = (float*)d_out;

    char* base = (char*)d_ws;
    size_t off = 0;
    auto alloc = [&](size_t bytes) { char* p = base + off; off += (bytes + 255) & ~(size_t)255; return p; };
    const size_t WSZ = (size_t)1024 * 256;
    _Float16* eWhh0f = (_Float16*)alloc(WSZ * 2);
    _Float16* eWih1f = (_Float16*)alloc(WSZ * 2);
    _Float16* eWhh1f = (_Float16*)alloc(WSZ * 2);
    _Float16* dWctxf = (_Float16*)alloc(WSZ * 2);
    _Float16* dWcurf = (_Float16*)alloc(1024 * 2 * 2);
    _Float16* dWhh0f = (_Float16*)alloc(WSZ * 2);
    _Float16* dWih1f = (_Float16*)alloc(WSZ * 2);
    _Float16* dWhh1f = (_Float16*)alloc(WSZ * 2);
    float* h0ws = (float*)alloc((size_t)NB * HDs * 4);
    float* c0ws = (float*)alloc((size_t)NB * HDs * 4);
    float* h1ws = (float*)alloc((size_t)NB * HDs * 4);
    float* c1ws = (float*)alloc((size_t)NB * HDs * 4);
    float* e1 = (float*)alloc((size_t)NB * 32 * 32 * 32 * 4);
    float* e2 = (float*)alloc((size_t)NB * 64 * 16 * 16 * 4);
    float* e3 = (float*)alloc((size_t)NB * 128 * 8 * 8 * 4);
    float* pooled = (float*)alloc((size_t)NB * 128 * 4);
    float* envf   = (float*)alloc((size_t)NB * 128 * 4);
    float* goalf  = (float*)alloc((size_t)NB * 64 * 4);
    float* ctxws  = (float*)alloc((size_t)NB * HDs * 4);
    (void)ws_size; (void)n_in; (void)in_sizes; (void)out_size;

    // weight conversion
    int wn = (int)WSZ;
    f32tof16_kernel<<<wn / 256, 256, 0, stream>>>(eWhh0, eWhh0f, wn);
    f32tof16_kernel<<<wn / 256, 256, 0, stream>>>(eWih1, eWih1f, wn);
    f32tof16_kernel<<<wn / 256, 256, 0, stream>>>(eWhh1, eWhh1f, wn);
    f32tof16_kernel<<<wn / 256, 256, 0, stream>>>(dWhh0, dWhh0f, wn);
    f32tof16_kernel<<<wn / 256, 256, 0, stream>>>(dWih1, dWih1f, wn);
    f32tof16_kernel<<<wn / 256, 256, 0, stream>>>(dWhh1, dWhh1f, wn);
    split_wih0_kernel<<<(1024 * 258 + 255) / 256, 256, 0, stream>>>(dWih0, dWctxf, dWcurf);

    // encoder
    encoder_kernel<<<NB / 16, 256, 0, stream>>>(hist, eWih0, ebih0, ebhh0, ebih1, ebhh1,
                                                eWhh0f, eWih1f, eWhh1f, h0ws, c0ws, h1ws, c1ws);

    // CNN path
    conv_s2_kernel<<<(NB * 32 * 32 * 32) / 256, 256, 0, stream>>>(env_map, c1w, c1b, e1, NB, 18, 64, 32, 32);
    conv_s2_kernel<<<(NB * 64 * 16 * 16) / 256, 256, 0, stream>>>(e1, c2w, c2b, e2, NB, 32, 32, 64, 16);
    conv_s2_kernel<<<(NB * 128 * 8 * 8) / 256, 256, 0, stream>>>(e2, c3w, c3b, e3, NB, 64, 16, 128, 8);
    pool_kernel<<<(NB * 128) / 256, 256, 0, stream>>>(e3, pooled);
    fc_kernel<<<(NB * 128) / 256, 256, 0, stream>>>(pooled, efcw, efcb, envf, NB, 128, 128, 0);
    fc_kernel<<<(NB * 64) / 256, 256, 0, stream>>>(goal, gfcw, gfcb, goalf, NB, 2, 64, 1);
    fusion_kernel<<<(NB * HDs) / 256, 256, 0, stream>>>(h1ws, envf, goalf, fw, fb, ctxws);

    // decoder
    decoder_kernel<<<NB / 16, 256, 0, stream>>>(hist, h0ws, c0ws, c1ws, ctxws,
                                                dbih0, dbhh0, dbih1, dbhh1,
                                                dWctxf, dWcurf, dWhh0f, dWih1f, dWhh1f,
                                                outw, outb, out);
}
